// MetricModel_57784490001154
// MI455X (gfx1250) — compile-verified, matrix operand order checked by reading, not run on hardware
//
#include <hip/hip_runtime.h>
#include <math.h>

#define HWSZ   9216     // 96*96
#define BSZ    32
#define CSZ    256
#define NITC   10
#define NCC    10
#define NROWS  3200     // NIT*NC*B
#define NKCH   288      // HWSZ/32
#define GAMMAF (0.3f / 9.0f)

typedef __attribute__((ext_vector_type(16))) __bf16 bf16x16;
typedef __attribute__((ext_vector_type(8)))  float  f32x8;

union V16 {
  bf16x16 v;
  unsigned short s[16];
  unsigned u[8];
  uint4 q[2];
};

static __device__ inline f32x8 wmma_bf16(bf16x16 a, bf16x16 b, f32x8 c) {
  // 8 args: (neg_a, A, neg_b, B, c_mod, C, reuse_a, reuse_b)
  return __builtin_amdgcn_wmma_f32_16x16x32_bf16(false, a, false, b, (short)0, c,
                                                 false, false);
}

// k-within-chunk coverage for lane-half g (matches 16-bit A VGPR layout)
static __device__ inline int ksel(int g, int j) {
  return (j < 8) ? (g * 8 + j) : (16 + g * 8 + (j - 8));
}

// ---------------------------------------------------------------------------
// Phase 1: pack label bits [9:0] and dropblock "keep" bits [25:16] per (b,hw)
// block[i,b,h,w] = 1 iff no noise<gamma in clamped 3x3 neighborhood (SAME pad)
// ---------------------------------------------------------------------------
__global__ __launch_bounds__(256) void masks_kernel(
    const float* __restrict__ noise,   // (NIT,B,96,96)
    const float* __restrict__ label,   // (B,12,96,96)
    unsigned* __restrict__ packed)     // (B*HWSZ)
{
  int idx = blockIdx.x * 256 + threadIdx.x;
  if (idx >= BSZ * HWSZ) return;
  int b  = idx / HWSZ;
  int hw = idx - b * HWSZ;
  int h  = hw / 96;
  int w  = hw - h * 96;

  unsigned lb = 0u;
#pragma unroll
  for (int p = 0; p < NCC; ++p) {
    float v = label[((size_t)b * 12 + p) * HWSZ + hw];
    lb |= (v != 0.0f) ? (1u << p) : 0u;
  }

  unsigned bb = 0u;
#pragma unroll
  for (int i = 0; i < NITC; ++i) {
    const float* np = noise + ((size_t)i * BSZ + b) * HWSZ;
    bool drop = false;
    for (int dh = -1; dh <= 1; ++dh) {
      int hh = h + dh;
      if (hh < 0 || hh >= 96) continue;
      for (int dw = -1; dw <= 1; ++dw) {
        int ww = w + dw;
        if (ww < 0 || ww >= 96) continue;
        drop = drop || (np[hh * 96 + ww] < GAMMAF);
      }
    }
    bb |= drop ? 0u : (1u << i);
  }
  packed[idx] = lb | (bb << 16);
}

// ---------------------------------------------------------------------------
// Phase 1b: expand mask bits into the A matrix in bf16, PRE-SWIZZLED into the
// exact per-lane WMMA operand layout:
//   aswz[((b*NKCH + kk)*7 + t)*32 + lane] -> 16 bf16 (32B contiguous)
// ---------------------------------------------------------------------------
__global__ __launch_bounds__(256) void aswizzle_kernel(
    const unsigned* __restrict__ packed,  // (B*HWSZ)
    unsigned* __restrict__ aswz)          // B*NKCH*7*32 * 8 u32
{
  size_t gid = (size_t)blockIdx.x * 256 + threadIdx.x;  // 32*288*7*32 threads
  int lane = (int)(gid & 31);
  size_t q = gid >> 5;
  int t = (int)(q % 7); q /= 7;
  int kk = (int)(q % NKCH); q /= NKCH;
  int b = (int)q;

  int g = lane >> 4, ln = lane & 15;
  int r = t * 16 + ln;
  unsigned valid = (r < 100) ? 1u : 0u;
  int i = valid ? (r / 10) : 0;
  int p = valid ? (r - 10 * i) : 0;

  const unsigned* wb = packed + (size_t)b * HWSZ + kk * 32;
  unsigned out[8];
#pragma unroll
  for (int jj = 0; jj < 8; ++jj) {
    unsigned w0 = wb[ksel(g, 2 * jj)];
    unsigned w1 = wb[ksel(g, 2 * jj + 1)];
    unsigned b0 = (w0 >> p) & (w0 >> (16 + i)) & valid & 1u;
    unsigned b1 = (w1 >> p) & (w1 >> (16 + i)) & valid & 1u;
    out[jj] = (b0 | (b1 << 16)) * 0x3F80u;  // two bf16 {0,1.0}
  }
  unsigned* dst = aswz + gid * 8;
  *(uint4*)(dst)     = make_uint4(out[0], out[1], out[2], out[3]);
  *(uint4*)(dst + 4) = make_uint4(out[4], out[5], out[6], out[7]);
}

// ---------------------------------------------------------------------------
// Phase 2: per-batch masked pooling as WMMA GEMM.
//   pooled[b, r=i*10+p, c] = sum_hw mask(i,p,b,hw) * feature[b,c,hw]
// ctile 0..15 -> 16 channels each; ctile 16 -> B==1.0 gives counts for free.
// All 7 A tiles batch-loaded into distinct registers, then a WMMA burst, so
// the compiler can clause the 14 b128 loads and stagger s_wait_loadcnt.
// ---------------------------------------------------------------------------
__global__ __launch_bounds__(32) void pool_wmma_kernel(
    const float* __restrict__ feature,    // (B,C,HWSZ)
    const unsigned* __restrict__ aswz,    // swizzled A
    float* __restrict__ pooled,           // (B,112,C)
    float* __restrict__ counts)           // (B,112)
{
  const int ctile = blockIdx.x;   // 0..16
  const int b     = blockIdx.y;   // 0..31
  const int lane  = threadIdx.x;  // 0..31
  const int g     = lane >> 4;
  const int ln    = lane & 15;
  const bool isCount = (ctile == 16);

  f32x8 acc[7] = {};
  const uint4* abase =
      (const uint4*)aswz + ((size_t)b * NKCH * 7 * 32 + lane) * 2;

  if (isCount) {
    V16 bu;
#pragma unroll
    for (int j = 0; j < 16; ++j) bu.s[j] = 0x3F80;  // bf16 1.0
    for (int kk = 0; kk < NKCH; ++kk) {
      const uint4* ak = abase + (size_t)kk * 7 * 64;
      V16 au[7];
#pragma unroll
      for (int t = 0; t < 7; ++t) {
        au[t].q[0] = ak[t * 64 + 0];
        au[t].q[1] = ak[t * 64 + 1];
      }
#pragma unroll
      for (int t = 0; t < 7; ++t) acc[t] = wmma_bf16(au[t].v, bu.v, acc[t]);
    }
    if (ln == 0) {
#pragma unroll
      for (int t = 0; t < 7; ++t)
#pragma unroll
        for (int v = 0; v < 8; ++v)
          counts[b * 112 + t * 16 + g * 8 + v] = acc[t][v];
    }
    return;
  }

  const float* fbase =
      feature + ((size_t)b * CSZ + (size_t)ctile * 16 + ln) * HWSZ;

  for (int kk = 0; kk < NKCH; ++kk) {
    const int hw0 = kk * 32 + g * 8;
    const int hw1 = hw0 + 16;
    float4 f0 = *(const float4*)(fbase + hw0);
    float4 f1 = *(const float4*)(fbase + hw0 + 4);
    float4 f2 = *(const float4*)(fbase + hw1);
    float4 f3 = *(const float4*)(fbase + hw1 + 4);

    const uint4* ak = abase + (size_t)kk * 7 * 64;
    V16 au[7];
#pragma unroll
    for (int t = 0; t < 7; ++t) {
      au[t].q[0] = ak[t * 64 + 0];
      au[t].q[1] = ak[t * 64 + 1];
    }

    float ff[16] = {f0.x, f0.y, f0.z, f0.w, f1.x, f1.y, f1.z, f1.w,
                    f2.x, f2.y, f2.z, f2.w, f3.x, f3.y, f3.z, f3.w};
    V16 bu;
#pragma unroll
    for (int j = 0; j < 16; ++j) bu.v[j] = (__bf16)ff[j];

    if (kk + 1 < NKCH)
      __builtin_prefetch((const void*)(fbase + hw0 + 32), 0, 3);

#pragma unroll
    for (int t = 0; t < 7; ++t) acc[t] = wmma_bf16(au[t].v, bu.v, acc[t]);
  }

  // C/D layout: lane ln = column, row m = g*8 + v
  const int c = ctile * 16 + ln;
#pragma unroll
  for (int t = 0; t < 7; ++t)
#pragma unroll
    for (int v = 0; v < 8; ++v) {
      int m = t * 16 + g * 8 + v;
      pooled[((size_t)b * 112 + m) * CSZ + c] = acc[t][v];
    }
}

// ---------------------------------------------------------------------------
// Phase 2b: feat = pooled/counts (safe), emit labels (dropblock scale cancels)
// ---------------------------------------------------------------------------
__global__ __launch_bounds__(256) void finalize_kernel(
    const float* __restrict__ pooled, const float* __restrict__ counts,
    float* __restrict__ feat, int* __restrict__ labels)
{
  int row = blockIdx.x;          // (i*NC + p)*B + b
  int c   = threadIdx.x;         // 0..255
  int i   = row / (NCC * BSZ);
  int rem = row - i * NCC * BSZ;
  int p   = rem / BSZ;
  int b   = rem - p * BSZ;
  int r   = i * 10 + p;
  float cnt = counts[b * 112 + r];
  float val = (cnt > 0.5f) ? pooled[((size_t)b * 112 + r) * CSZ + c] / cnt : 0.0f;
  feat[(size_t)row * CSZ + c] = val;
  if (c == 0) labels[row] = p;
}

// ---------------------------------------------------------------------------
// Phase 3: fused MLP. 16 rows per block, 8 waves. Input rows staged into LDS
// via CDNA5 async global->LDS copies (ASYNCcnt), then three bf16 WMMA GEMMs.
// ---------------------------------------------------------------------------
__global__ __launch_bounds__(256) void mlp_kernel(
    const float* __restrict__ feat, const float* __restrict__ W1,
    const float* __restrict__ b1, const float* __restrict__ W2,
    const float* __restrict__ b2, const float* __restrict__ Wc,
    float* __restrict__ out_f2n, float* __restrict__ out_logit)
{
  __shared__ float  s_xf[16 * 256];   // async staging (f32)
  __shared__ __bf16 s_x[16 * 256];
  __shared__ __bf16 s_h[16 * 128];
  __shared__ float  s_f2[16 * 128];
  __shared__ __bf16 s_f2b[16 * 128];
  __shared__ float  s_nrm[16];

  const int tid  = threadIdx.x;
  const int wave = tid >> 5;
  const int lane = tid & 31;
  const int g    = lane >> 4;
  const int ln   = lane & 15;
  const int row0 = blockIdx.x * 16;

  // --- async copy 16x256 f32 block into LDS (4 x b128 per thread) ---
  for (int e = tid; e < 1024; e += 256) {
    unsigned long long ga = (unsigned long long)(uintptr_t)(
        feat + (size_t)(row0 + (e >> 6)) * 256 + (size_t)(e & 63) * 4);
    // generic LDS pointers carry the LDS byte offset in their low 32 bits
    unsigned la = (unsigned)(uintptr_t)&s_xf[e * 4];
    asm volatile("global_load_async_to_lds_b128 %0, %1, off"
                 :: "v"(la), "v"(ga) : "memory");
  }
  asm volatile("s_wait_asynccnt 0" ::: "memory");
  __syncthreads();
  for (int e = tid; e < 16 * 256; e += 256) s_x[e] = (__bf16)s_xf[e];
  __syncthreads();

  {  // GEMM1: h = relu(x @ W1 + b1)
    const int nb = wave * 16 + ln;
    f32x8 acc = {};
    for (int kk = 0; kk < 8; ++kk) {
      const int k0 = kk * 32 + g * 8;
      const int k1 = k0 + 16;
      bf16x16 a, bm;
#pragma unroll
      for (int j = 0; j < 8; ++j) {
        a[j]      = s_x[ln * 256 + k0 + j];
        a[8 + j]  = s_x[ln * 256 + k1 + j];
        bm[j]     = (__bf16)W1[(size_t)(k0 + j) * 128 + nb];
        bm[8 + j] = (__bf16)W1[(size_t)(k1 + j) * 128 + nb];
      }
      acc = wmma_bf16(a, bm, acc);
    }
    const float bias = b1[nb];
#pragma unroll
    for (int v = 0; v < 8; ++v) {
      int m = g * 8 + v;
      s_h[m * 128 + nb] = (__bf16)fmaxf(acc[v] + bias, 0.0f);
    }
  }
  __syncthreads();

  {  // GEMM2: f2 = h @ W2 + b2
    const int nb = wave * 16 + ln;
    f32x8 acc = {};
    for (int kk = 0; kk < 4; ++kk) {
      const int k0 = kk * 32 + g * 8;
      const int k1 = k0 + 16;
      bf16x16 a, bm;
#pragma unroll
      for (int j = 0; j < 8; ++j) {
        a[j]      = s_h[ln * 128 + k0 + j];
        a[8 + j]  = s_h[ln * 128 + k1 + j];
        bm[j]     = (__bf16)W2[(size_t)(k0 + j) * 128 + nb];
        bm[8 + j] = (__bf16)W2[(size_t)(k1 + j) * 128 + nb];
      }
      acc = wmma_bf16(a, bm, acc);
    }
    const float bias = b2[nb];
#pragma unroll
    for (int v = 0; v < 8; ++v) {
      int m = g * 8 + v;
      float fv = acc[v] + bias;
      s_f2[m * 128 + nb]  = fv;
      s_f2b[m * 128 + nb] = (__bf16)fv;
    }
  }
  __syncthreads();

  if (wave == 0) {  // GEMM3: logit = f2 @ Wc (Wc cols padded 10->16)
    f32x8 acc = {};
    for (int kk = 0; kk < 4; ++kk) {
      const int k0 = kk * 32 + g * 8;
      const int k1 = k0 + 16;
      bf16x16 a, bm;
#pragma unroll
      for (int j = 0; j < 8; ++j) {
        a[j]      = s_f2b[ln * 128 + k0 + j];
        a[8 + j]  = s_f2b[ln * 128 + k1 + j];
        bm[j]     = (ln < 10) ? (__bf16)Wc[(size_t)(k0 + j) * 10 + ln] : (__bf16)0.0f;
        bm[8 + j] = (ln < 10) ? (__bf16)Wc[(size_t)(k1 + j) * 10 + ln] : (__bf16)0.0f;
      }
      acc = wmma_bf16(a, bm, acc);
    }
    if (ln < 10) {
#pragma unroll
      for (int v = 0; v < 8; ++v)
        out_logit[(size_t)(row0 + g * 8 + v) * 10 + ln] = acc[v];
    }
  }

  if (tid < 16) {
    float s = 0.0f;
    for (int k = 0; k < 128; ++k) { float x = s_f2[tid * 128 + k]; s += x * x; }
    s_nrm[tid] = fmaxf(sqrtf(s), 1e-12f);
  }
  __syncthreads();

  for (int e = tid; e < 16 * 128; e += 256)
    out_f2n[(size_t)(row0 + (e >> 7)) * 128 + (e & 127)] = s_f2[e] / s_nrm[e >> 7];
}

// ---------------------------------------------------------------------------
extern "C" void kernel_launch(void* const* d_in, const int* in_sizes, int n_in,
                              void* d_out, int out_size, void* d_ws, size_t ws_size,
                              hipStream_t stream) {
  const float* feature    = (const float*)d_in[0];
  const float* label_lst  = (const float*)d_in[1];
  const float* drop_noise = (const float*)d_in[2];
  const float* W1         = (const float*)d_in[3];
  const float* b1         = (const float*)d_in[4];
  const float* W2         = (const float*)d_in[5];
  const float* b2         = (const float*)d_in[6];
  const float* Wc         = (const float*)d_in[7];
  float* out = (float*)d_out;

  // workspace (u32/f32 elems): packed 294912 | aswz 16515072 | pooled 917504
  //                            counts 3584 | feat 819200  (~74 MB total)
  float* wsf       = (float*)d_ws;
  unsigned* packed = (unsigned*)wsf;
  unsigned* aswz   = (unsigned*)(wsf + BSZ * HWSZ);
  float* pooled    = wsf + BSZ * HWSZ + (size_t)BSZ * NKCH * 7 * 32 * 8;
  float* counts    = pooled + BSZ * 112 * CSZ;
  float* feat      = counts + BSZ * 112;

  masks_kernel<<<(BSZ * HWSZ + 255) / 256, 256, 0, stream>>>(drop_noise, label_lst,
                                                             packed);
  aswizzle_kernel<<<(BSZ * NKCH * 7 * 32) / 256, 256, 0, stream>>>(packed, aswz);

  dim3 g2(17, BSZ);
  pool_wmma_kernel<<<g2, 32, 0, stream>>>(feature, aswz, pooled, counts);

  // d_out layout: [f2/||f2|| : 3200*128][labels(i32): 3200][logit : 3200*10]
  finalize_kernel<<<NROWS, 256, 0, stream>>>(pooled, counts, feat,
                                             (int*)(out + 3200 * 128));
  mlp_kernel<<<NROWS / 16, 256, 0, stream>>>(feat, W1, b1, W2, b2, Wc, out,
                                             out + 3200 * 128 + 3200);
}
